// RNN_44908178047035
// MI455X (gfx1250) — matrix operand order repaired and checked by reading, not compiled, not run on hardware
//
#include <hip/hip_runtime.h>

// ---------------------------------------------------------------------------
// Persistent wave32 WMMA RNN for gfx1250 (MI455X).
//   T=1024, B=256, NI=6, H=512
//   drive = inputs @ W_ih + b_h + sigma*noise
//   state = (1-a)*state + a*(relu(state) @ W_hh + drive)
//
// 16 persistent workgroups; each owns 16 batch rows for all T steps.
// Per step per WG: (16 x 544) bf16 A  x  (544 x 512) bf16 B, fp32 accumulate.
// B-operand (weights) is served from three tiers to balance the WGP's two
// ~256B/clk operand paths:
//   kb 0..5   -> persistently held in VGPRs   (96 VGPRs/wave, zero traffic)
//   kb 6..8   -> LDS cache (96 KB, filled once, ds path)
//   kb 9..16  -> streamed from L2 each step   (vmem path, anti-hoisted)
// ---------------------------------------------------------------------------

#define T_STEPS 1024
#define BATCH   256
#define NIN     6
#define HID     512
#define KBLK    17          // 16 K-blocks of state (512) + 1 block for inputs
#define NT      32          // 512 / 16 output column tiles
#define FRAG_US 512         // ushorts per packed fragment (32 lanes * 16)

#define KB_REG  6                    // kb 0..5 in registers
#define KB_LDS  3                    // kb 6..8 in LDS
#define KB_LDS0 (KB_REG)             // = 6
#define KB_GLB0 (KB_REG + KB_LDS)    // = 9, kb 9..16 streamed

static constexpr float ALPHA = 0.2f;
static constexpr float SIGMA = 0.15811388300841897f; // sqrt(2/0.2)*0.05

typedef __attribute__((ext_vector_type(16))) __bf16 v16bf;
typedef __attribute__((ext_vector_type(8)))  float  v8f;

__device__ __forceinline__ unsigned short f2bf(float f) {
    // round-to-nearest-even fp32 -> bf16
    unsigned int u = __float_as_uint(f);
    unsigned int r = (u + 0x7FFFu + ((u >> 16) & 1u)) >> 16;
    return (unsigned short)r;
}

// ---------------------------------------------------------------------------
// Pre-pack W_hh (512x512 f32, row-major [K][N]) and W_ih (6x512) into bf16
// B-fragment images: fragment (nt, kb) holds B-block K=[kb*32,+32), N=[nt*16,+16)
// Contiguous B layout used here (16-bit, 32x16): lanes 0-15 = col n, K 0..15; lanes 16-31 = K 16..31.
// Stored lane-major: wpack[(nt*17+kb)*512 + lane*16 + j].
// ---------------------------------------------------------------------------
__global__ void rnn_pack_w(const float* __restrict__ whh,
                           const float* __restrict__ wih,
                           unsigned short* __restrict__ wpack) {
    int gid  = blockIdx.x * blockDim.x + threadIdx.x;   // < 544*32
    int lane = gid & 31;
    int fid  = gid >> 5;                                 // 0..543
    int nt   = fid / KBLK;
    int kb   = fid - nt * KBLK;
    int n    = nt * 16 + (lane & 15);
    int half = lane >> 4;
    unsigned short* dst = wpack + (size_t)fid * FRAG_US + lane * 16;
    if (kb < 16) {
        int k0 = kb * 32 + half * 16;
        #pragma unroll
        for (int j = 0; j < 16; ++j)
            dst[j] = f2bf(whh[(k0 + j) * HID + n]);
    } else {
        // input-projection block: rows K=0..5 are W_ih, rest zero-padded
        #pragma unroll
        for (int j = 0; j < 16; ++j) {
            float v = (half == 0 && j < NIN) ? wih[j * HID + n] : 0.0f;
            dst[j] = f2bf(v);
        }
    }
}

// ---------------------------------------------------------------------------
// Persistent recurrent kernel: 16 blocks x 512 threads (16 wave32s).
// Wave w computes output column tiles nt = {2w, 2w+1}.
// ---------------------------------------------------------------------------
__global__ void __launch_bounds__(512)
rnn_persist(const float* __restrict__ inputs,   // [T,B,NI]
            const float* __restrict__ init,     // [B,H]
            const float* __restrict__ bh,       // [1,H]
            const float* __restrict__ noise,    // [T,B,H]
            const unsigned short* __restrict__ wpack,
            float* __restrict__ out) {          // [T+1,B,H]
    // Double-buffered bf16 A-fragment image: [buf][kb][lane][32B]   (34 KB)
    __shared__ __align__(32) unsigned char aimg[2][KBLK * 1024];
    // LDS-resident B fragments for kb in [KB_LDS0, KB_GLB0):        (96 KB)
    // layout: [nt][kbl][lane][32B]
    __shared__ __align__(32) unsigned char bcache[NT * KB_LDS * 1024];

    const int tid   = threadIdx.x;
    const int lane  = tid & 31;
    const int wave  = tid >> 5;
    const int laneN = lane & 15;
    const int half  = lane >> 4;
    const int b0    = blockIdx.x * 16;
    const int nt0   = wave * 2;
    const int c0    = nt0 * 16 + laneN;
    const int c1    = c0 + 16;
    const float bh0 = bh[c0];
    const float bh1 = bh[c1];

    // ---- fill the LDS B-cache (once) ----
    // 96 fragments * 1KB = 6144 16-byte chunks, 12 per thread
    for (int j = tid; j < NT * KB_LDS * 64; j += 512) {
        int f   = j >> 6;               // frag = nt*KB_LDS + kbl
        int e   = j & 63;               // 16B chunk within fragment
        int nt  = f / KB_LDS;
        int kbl = f - nt * KB_LDS;
        ((uint4*)bcache)[j] =
            ((const uint4*)wpack)[(nt * KBLK + KB_LDS0 + kbl) * 64 + e];
    }

    // Write relu(state) element (row m, state-dim c) into A image buffer `buf`
    // contiguous A layout (matches B packing): lanes 0-15 hold K 0..15, lanes 16-31 K 16..31.
    auto put = [&](int buf, int m, int c, float v) {
        int kb  = c >> 5;
        int kk  = c & 31;
        int dl  = kk >> 4;                        // destination lane half
        int pos = kk & 15;                        // halfword 0..15 within lane
        *(unsigned short*)(&aimg[buf][kb * 1024 + (dl * 16 + m) * 32 + pos * 2])
            = f2bf(v > 0.0f ? v : 0.0f);
    };

    // ---- init: state registers, out[0], A image buffer 0 ----
    float st0[8], st1[8];
    #pragma unroll
    for (int r = 0; r < 8; ++r) {
        int m = half * 8 + r;
        st0[r] = init[(b0 + m) * HID + c0];
        st1[r] = init[(b0 + m) * HID + c1];
        out[(long)(b0 + m) * HID + c0] = st0[r];
        out[(long)(b0 + m) * HID + c1] = st1[r];
        put(0, m, c0, st0[r]);
        put(0, m, c1, st1[r]);
    }
    if (tid < 256) { // zero the padded input slot (kb=16) of both buffers
        ((unsigned int*)(&aimg[0][16 * 1024]))[tid] = 0u;
        ((unsigned int*)(&aimg[1][16 * 1024]))[tid] = 0u;
    }
    __syncthreads();
    if (tid < 16) {  // stage inputs[0] into input slot of buffer 0 (K=0..5)
        const float* ip = inputs + (long)(b0 + tid) * NIN;
        unsigned short* d = (unsigned short*)(&aimg[0][16 * 1024 + tid * 32]);
        #pragma unroll
        for (int i = 0; i < NIN; ++i) d[i] = f2bf(ip[i]);
    }
    __syncthreads();

    // ---- register-resident B fragments: kb 0..KB_REG-1, both N-tiles ----
    const unsigned short* wb0 = wpack + (size_t)(nt0    ) * KBLK * FRAG_US + lane * 16;
    const unsigned short* wb1 = wpack + (size_t)(nt0 + 1) * KBLK * FRAG_US + lane * 16;
    v16bf breg[2 * KB_REG];
    #pragma unroll
    for (int k = 0; k < KB_REG; ++k) {
        breg[2 * k]     = *(const v16bf*)(wb0 + k * FRAG_US);
        breg[2 * k + 1] = *(const v16bf*)(wb1 + k * FRAG_US);
    }

    // Scalar bases for the streamed tier (kb >= KB_GLB0), byte units
    const unsigned char* gb0 =
        (const unsigned char*)(wpack + (size_t)(nt0    ) * KBLK * FRAG_US) + KB_GLB0 * 1024;
    const unsigned char* gb1 =
        (const unsigned char*)(wpack + (size_t)(nt0 + 1) * KBLK * FRAG_US) + KB_GLB0 * 1024;
    const unsigned char* lb0 = &bcache[(nt0    ) * KB_LDS * 1024];
    const unsigned char* lb1 = &bcache[(nt0 + 1) * KB_LDS * 1024];

    // Loop-carried opaque per-lane byte offset: prevents the compiler from
    // proving the streamed B loads loop-invariant (round-1 bug: it hoisted
    // all weights into registers and spilled them to scratch).
    unsigned ovoff = (unsigned)lane * 32u;

    for (int t = 0; t < T_STEPS; ++t) {
        const int cur = t & 1;
        const int nxt = cur ^ 1;
        asm volatile("" : "+v"(ovoff));   // opaque: re-materialize each step

        // ---- WMMA chain: acc = relu(state) @ W_hh + inputs @ W_ih ----
        v8f acc0 = {};
        v8f acc1 = {};
        #pragma unroll
        for (int kb = 0; kb < KBLK; ++kb) {
            v16bf a = *(const v16bf*)(&aimg[cur][kb * 1024] + ovoff);
            v16bf bA, bB;
            if (kb < KB_REG) {                 // tier 1: registers
                bA = breg[2 * kb];
                bB = breg[2 * kb + 1];
            } else if (kb < KB_GLB0) {         // tier 2: LDS cache
                bA = *(const v16bf*)(lb0 + (kb - KB_LDS0) * 1024 + ovoff);
                bB = *(const v16bf*)(lb1 + (kb - KB_LDS0) * 1024 + ovoff);
            } else {                           // tier 3: streamed from L2
                bA = *(const v16bf*)(gb0 + (kb - KB_GLB0) * 1024 + ovoff);
                bB = *(const v16bf*)(gb1 + (kb - KB_GLB0) * 1024 + ovoff);
            }
            acc0 = __builtin_amdgcn_wmma_f32_16x16x32_bf16(
                       false, a, false, bA, (short)0, acc0, false, false);
            acc1 = __builtin_amdgcn_wmma_f32_16x16x32_bf16(
                       false, a, false, bB, (short)0, acc1, false, false);
        }

        // ---- leaky-integrator update, store out[t+1], refill image[nxt] ----
        const float* np = noise + ((long)t * BATCH + b0) * HID;
        float*       op = out   + ((long)(t + 1) * BATCH + b0) * HID;
        #pragma unroll
        for (int r = 0; r < 8; ++r) {
            int m = half * 8 + r;
            float n0 = np[m * HID + c0];
            float n1 = np[m * HID + c1];
            float sn0 = acc0[r] + bh0 + SIGMA * n0;
            float sn1 = acc1[r] + bh1 + SIGMA * n1;
            st0[r] = (1.0f - ALPHA) * st0[r] + ALPHA * sn0;
            st1[r] = (1.0f - ALPHA) * st1[r] + ALPHA * sn1;
            op[m * HID + c0] = st0[r];
            op[m * HID + c1] = st1[r];
            put(nxt, m, c0, st0[r]);
            put(nxt, m, c1, st1[r]);
        }
        if (t + 1 < T_STEPS && tid < 16) {
            const float* ip = inputs + ((long)(t + 1) * BATCH + b0 + tid) * NIN;
            unsigned short* d = (unsigned short*)(&aimg[nxt][16 * 1024 + tid * 32]);
            #pragma unroll
            for (int i = 0; i < NIN; ++i) d[i] = f2bf(ip[i]);
        }
        __syncthreads();   // image[nxt] complete before next step's WMMA reads
    }
}

extern "C" void kernel_launch(void* const* d_in, const int* in_sizes, int n_in,
                              void* d_out, int out_size, void* d_ws, size_t ws_size,
                              hipStream_t stream) {
    const float* inputs = (const float*)d_in[0];   // [T,B,NI]
    const float* init   = (const float*)d_in[1];   // [B,H]
    const float* wih    = (const float*)d_in[2];   // [NI,H]
    const float* whh    = (const float*)d_in[3];   // [H,H]
    const float* bh     = (const float*)d_in[4];   // [1,H]
    const float* noise  = (const float*)d_in[5];   // [T,B,H]
    float* out = (float*)d_out;
    unsigned short* wpack = (unsigned short*)d_ws; // needs 544*512*2 = 557 KB

    // Pack weights into WMMA B-fragment images (544 fragments * 32 lanes).
    rnn_pack_w<<<68, 256, 0, stream>>>(whh, wih, wpack);

    // Persistent recurrent kernel: 16 batch tiles, 512 threads (16 wave32s) each.
    rnn_persist<<<16, 512, 0, stream>>>(inputs, init, bh, noise, wpack, out);
}
